// DaGMM_23072564314153
// MI455X (gfx1250) — matrix operand order repaired and checked by reading, not compiled, not run on hardware
//
#include <hip/hip_runtime.h>
#include <hip/hip_bf16.h>

// ---------------------------------------------------------------------------
// DaGMM forward for MI455X (gfx1250, wave32, WMMA).
// Dominant cost: 3 streams of the 1 GB f32 adjacency -> bf16 WMMA, f32 acc.
// Ping-pong LDS double buffering, one barrier per K-step, templated epilogue.
// ---------------------------------------------------------------------------

typedef __attribute__((ext_vector_type(16))) __bf16 v16bf;
typedef __attribute__((ext_vector_type(8)))  float  v8f;

#define NN      16384   // nodes
#define NG      512     // graphs
#define IND     512     // input dim
#define H1D     128
#define H2D     32
#define LATP    16      // latent 4 padded to 16 for WMMA N-tile
#define NGMM    10

#define BM      128     // C rows per block
#define KB      32      // K step (bf16 WMMA K)
#define ASTR    40      // LDS A row stride in ushorts (padded, keeps 16B align)

__device__ __forceinline__ unsigned short f2bf(float f) {
  unsigned int u = __float_as_uint(f);
  unsigned int r = u + 0x7FFFu + ((u >> 16) & 1u);   // round-to-nearest-even
  return (unsigned short)(r >> 16);
}

// ---------------------------------------------------------------------------
// Generic GEMM: C[M,N] = act(A[M,K](f32) * B[K,N](bf16) + bias)
// 256 threads = 8 wave32s; wave w owns the 16x16 C tile at rows bm+16w, col n0.
// A converted f32->bf16 on the fly into double-buffered LDS.
// Requires M%128==0, K%32==0, N%16==0 (true for all call sites here).
// ---------------------------------------------------------------------------
template <bool HASBIAS, bool RELU, bool OUTBF>
__global__ __launch_bounds__(256) void gemm_bf16wmma(
    const float* __restrict__ A, const unsigned short* __restrict__ B,
    void* __restrict__ Cout, const float* __restrict__ bias,
    int M, int N, int K)
{
  __shared__ unsigned short lA[2][BM * ASTR];   // 128 x 32 bf16, padded stride
  __shared__ unsigned short lB[2][16 * KB];     // B tile transposed: Bt[n][k]

  const int  t    = threadIdx.x;
  const int  lane = t & 31;
  const int  wave = t >> 5;                     // 0..7 -> m-tile within block
  const long bm   = (long)blockIdx.x * BM;
  const int  n0   = blockIdx.y * 16;

  v8f acc = {0.f, 0.f, 0.f, 0.f, 0.f, 0.f, 0.f, 0.f};

  // Fragment addressing (ISA 7.12.2, 16-bit A 16x32 / B 32x16 layouts)
  const int mloc  = wave * 16 + (lane & 15);
  const int kb    = (lane < 16) ? 0 : 8;        // A: K base for this lane half
  const int nn    = lane & 15;                  // B: column
  const int kbase = (lane < 16) ? 0 : 8;        // B: K base for this lane half

  // per-thread staging registers for the pipelined tile
  float4         fa[4];
  unsigned short bs[2];

  // cooperative global fetch of tile at k0 into registers
  auto gfetch = [&](int k0) {
#pragma unroll
    for (int i = 0; i < 4; ++i) {
      int q   = t + 256 * i;                    // 0..1023 chunk id
      int row = q >> 3;                         // 0..127
      int c4  = (q & 7) * 4;                    // 0..28
      fa[i] = *(const float4*)(A + (bm + row) * (long)K + k0 + c4);
    }
#pragma unroll
    for (int i = 0; i < 2; ++i) {
      int idx = t * 2 + i;                      // 0..511
      int kk  = idx >> 4;                       // 0..31
      int cc  = idx & 15;                       // 0..15
      bs[i] = B[(long)(k0 + kk) * N + n0 + cc];
    }
  };
  // convert + store staged tile into LDS buffer `buf`
  auto lcommit = [&](int buf) {
#pragma unroll
    for (int i = 0; i < 4; ++i) {
      int q   = t + 256 * i;
      int row = q >> 3;
      int c4  = (q & 7) * 4;
      ushort4 h;
      h.x = f2bf(fa[i].x); h.y = f2bf(fa[i].y);
      h.z = f2bf(fa[i].z); h.w = f2bf(fa[i].w);
      *(ushort4*)(&lA[buf][row * ASTR + c4]) = h;   // ds_store_b64
    }
#pragma unroll
    for (int i = 0; i < 2; ++i) {
      int idx = t * 2 + i;
      int kk  = idx >> 4;
      int cc  = idx & 15;
      lB[buf][cc * KB + kk] = bs[i];
    }
  };

  const int nsteps = K / KB;
  gfetch(0);
  lcommit(0);
  __syncthreads();

  for (int s = 0; s < nsteps; ++s) {
    const int buf = s & 1;
    if (s + 1 < nsteps) {
      gfetch((s + 1) * KB);                     // loads in flight over the WMMA
      if (s + 2 < nsteps)                       // gfx1250 global_prefetch_b8
        __builtin_prefetch(A + (bm + (t >> 1)) * (long)K + (s + 2) * KB, 0, 3);
    }

    // build fragments from LDS (16B-aligned b128 reads) and WMMA
    union { uint4 q[2]; v16bf v; } af, bfr;
    af.q[0]  = *(const uint4*)(&lA[buf][mloc * ASTR + kb]);       // K kb..+7
    af.q[1]  = *(const uint4*)(&lA[buf][mloc * ASTR + kb + 16]);  // K kb+16..+23
    bfr.q[0] = *(const uint4*)(&lB[buf][nn * KB + kbase]);        // K kbase..+7
    bfr.q[1] = *(const uint4*)(&lB[buf][nn * KB + kbase + 16]);   // K kbase+16..+23

    acc = __builtin_amdgcn_wmma_f32_16x16x32_bf16(
        false, af.v, false, bfr.v, (short)0, acc, false, false);

    if (s + 1 < nsteps) lcommit(buf ^ 1);
    __syncthreads();                            // single barrier per K-step
  }

  // straight-line epilogue (templated: no runtime branches)
  const int  colg  = n0 + (lane & 15);
  const long rbase = bm + wave * 16 + ((lane >> 4) * 8);
  float bv = 0.f;
  if (HASBIAS) bv = bias[colg];
#pragma unroll
  for (int r = 0; r < 8; ++r) {
    float v = acc[r] + bv;
    if (RELU) v = fmaxf(v, 0.f);
    long off = (rbase + r) * (long)N + colg;
    if (OUTBF) ((unsigned short*)Cout)[off] = f2bf(v);
    else       ((float*)Cout)[off] = v;
  }
}

// f32 [rows, cs] -> bf16 [rows, cd], zero-padded columns (for w3: 4 -> 16)
__global__ void cvt_pad_bf16(const float* __restrict__ src,
                             unsigned short* __restrict__ dst,
                             int rows, int cs, int cd)
{
  int i = blockIdx.x * blockDim.x + threadIdx.x;
  if (i >= rows * cd) return;
  int r = i / cd, c = i % cd;
  dst[i] = (c < cs) ? f2bf(src[r * cs + c]) : (unsigned short)0;
}

// b3 [4] -> f32 [16] zero padded
__global__ void pad_bias16(const float* __restrict__ b, float* __restrict__ o)
{
  int i = threadIdx.x;
  if (i < 16) o[i] = (i < 4) ? b[i] : 0.f;
}

// per-graph mean of enc[:, 0:4]  (enc is [NN, LATP] f32); one wave per graph
__global__ __launch_bounds__(32) void seg_mean(
    const float* __restrict__ enc, const int* __restrict__ bounds,
    float* __restrict__ out)
{
  int g     = blockIdx.x;
  int start = (g == 0) ? 0 : bounds[g - 1];
  int end   = bounds[g];
  int lane  = threadIdx.x;
  float sx = 0.f, sy = 0.f, sz = 0.f, sw = 0.f;
  for (int i = start + lane; i < end; i += 32) {
    const float4 v = *(const float4*)(enc + (long)i * LATP);
    sx += v.x; sy += v.y; sz += v.z; sw += v.w;
  }
#pragma unroll
  for (int off = 16; off > 0; off >>= 1) {
    sx += __shfl_down(sx, off);
    sy += __shfl_down(sy, off);
    sz += __shfl_down(sz, off);
    sw += __shfl_down(sw, off);
  }
  if (lane == 0) {
    float inv = 1.f / (float)(end - start);
    out[g * 4 + 0] = sx * inv;
    out[g * 4 + 1] = sy * inv;
    out[g * 4 + 2] = sz * inv;
    out[g * 4 + 3] = sw * inv;
  }
}

// estimation net: e = relu(out@ew1+eb1); gamma = softmax(e@ew2+eb2)
__global__ __launch_bounds__(256) void estimate(
    const float* __restrict__ pooled, const float* __restrict__ ew1,
    const float* __restrict__ eb1,    const float* __restrict__ ew2,
    const float* __restrict__ eb2,    float* __restrict__ gamma)
{
  int g = blockIdx.x * 256 + threadIdx.x;
  if (g >= NG) return;
  float o0 = pooled[g * 4 + 0], o1 = pooled[g * 4 + 1];
  float o2 = pooled[g * 4 + 2], o3 = pooled[g * 4 + 3];
  float e[H2D];
#pragma unroll
  for (int j = 0; j < H2D; ++j) {
    float v = eb1[j] + o0 * ew1[0 * H2D + j] + o1 * ew1[1 * H2D + j]
                     + o2 * ew1[2 * H2D + j] + o3 * ew1[3 * H2D + j];
    e[j] = fmaxf(v, 0.f);
  }
  float lg[NGMM];
  float mx = -3.4e38f;
#pragma unroll
  for (int c = 0; c < NGMM; ++c) {
    float v = eb2[c];
    for (int j = 0; j < H2D; ++j) v += e[j] * ew2[j * NGMM + c];
    lg[c] = v;
    mx = fmaxf(mx, v);
  }
  float s = 0.f;
#pragma unroll
  for (int c = 0; c < NGMM; ++c) { lg[c] = __expf(lg[c] - mx); s += lg[c]; }
  float inv = 1.f / s;
#pragma unroll
  for (int c = 0; c < NGMM; ++c) gamma[g * NGMM + c] = lg[c] * inv;
}

extern "C" void kernel_launch(void* const* d_in, const int* in_sizes, int n_in,
                              void* d_out, int out_size, void* d_ws, size_t ws_size,
                              hipStream_t stream) {
  const float* x   = (const float*)d_in[0];
  const float* adj = (const float*)d_in[1];
  const int*   gb  = (const int*)d_in[2];
  const float* w1  = (const float*)d_in[3];
  const float* b1  = (const float*)d_in[4];
  const float* w2  = (const float*)d_in[5];
  const float* b2  = (const float*)d_in[6];
  const float* w3  = (const float*)d_in[7];
  const float* b3  = (const float*)d_in[8];
  const float* ew1 = (const float*)d_in[9];
  const float* eb1 = (const float*)d_in[10];
  const float* ew2 = (const float*)d_in[11];
  const float* eb2 = (const float*)d_in[12];
  float* out = (float*)d_out;

  // workspace carve-up (256B aligned), total ~17.5 MB
  char* ws = (char*)d_ws;
  size_t off = 0;
  auto carve = [&](size_t bytes) -> void* {
    void* p = ws + off;
    off = (off + bytes + 255) & ~(size_t)255;
    return p;
  };
  unsigned short* w1b  = (unsigned short*)carve((size_t)IND * H1D * 2);
  unsigned short* w2b  = (unsigned short*)carve((size_t)H1D * H2D * 2);
  unsigned short* w3b  = (unsigned short*)carve((size_t)H2D * LATP * 2);
  float*          b3p  = (float*)carve(LATP * 4);
  unsigned short* xw1b = (unsigned short*)carve((size_t)NN * H1D * 2);
  unsigned short* t2b  = (unsigned short*)carve((size_t)NN * H2D * 2);
  unsigned short* t3b  = (unsigned short*)carve((size_t)NN * LATP * 2);
  float*          h1   = (float*)carve((size_t)NN * H1D * 4);
  float*          h2   = (float*)carve((size_t)NN * H2D * 4);
  float*          enc  = (float*)carve((size_t)NN * LATP * 4);

  // output layout: x [NN*IND] | pooled [NG*4] | gamma [NG*NGMM]
  float* out_pooled = out + (size_t)NN * IND;
  float* out_gamma  = out_pooled + (size_t)NG * 4;

  // x passthrough (first tuple element)
  hipMemcpyAsync(d_out, (const void*)x, (size_t)NN * IND * sizeof(float),
                 hipMemcpyDeviceToDevice, stream);

  // weight conversions
  cvt_pad_bf16<<<(IND * H1D + 255) / 256, 256, 0, stream>>>(w1, w1b, IND, H1D, H1D);
  cvt_pad_bf16<<<(H1D * H2D + 255) / 256, 256, 0, stream>>>(w2, w2b, H1D, H2D, H2D);
  cvt_pad_bf16<<<(H2D * LATP + 255) / 256, 256, 0, stream>>>(w3, w3b, H2D, 4, LATP);
  pad_bias16<<<1, 16, 0, stream>>>(b3, b3p);

  dim3 blk(256);
  // layer 1: xw1 = x @ w1 (bf16 out), then h1 = relu(adj @ xw1 + b1)
  gemm_bf16wmma<false, false, true><<<dim3(NN / BM, H1D / 16), blk, 0, stream>>>(
      x, w1b, xw1b, nullptr, NN, H1D, IND);
  gemm_bf16wmma<true, true, false><<<dim3(NN / BM, H1D / 16), blk, 0, stream>>>(
      adj, xw1b, h1, b1, NN, H1D, NN);
  // layer 2: t2 = h1 @ w2 (bf16 out), h2 = relu(adj @ t2 + b2)
  gemm_bf16wmma<false, false, true><<<dim3(NN / BM, H2D / 16), blk, 0, stream>>>(
      h1, w2b, t2b, nullptr, NN, H2D, H1D);
  gemm_bf16wmma<true, true, false><<<dim3(NN / BM, H2D / 16), blk, 0, stream>>>(
      adj, t2b, h2, b2, NN, H2D, NN);
  // layer 3: t3 = h2 @ w3pad (bf16), enc = adj @ t3 + b3pad
  gemm_bf16wmma<false, false, true><<<dim3(NN / BM, LATP / 16), blk, 0, stream>>>(
      h2, w3b, t3b, nullptr, NN, LATP, H2D);
  gemm_bf16wmma<true, false, false><<<dim3(NN / BM, LATP / 16), blk, 0, stream>>>(
      adj, t3b, enc, b3p, NN, LATP, NN);

  // pooling + estimation net
  seg_mean<<<NG, 32, 0, stream>>>(enc, gb, out_pooled);
  estimate<<<(NG + 255) / 256, 256, 0, stream>>>(out_pooled, ew1, eb1, ew2, eb2,
                                                 out_gamma);
}
